// GCNConv_36223754174749
// MI455X (gfx1250) — matrix operand order repaired and checked by reading, not compiled, not run on hardware
//
#include <hip/hip_runtime.h>
#include <stdint.h>

// ---- raw clang vector types (trivially-copyable, union-safe) ----
typedef __bf16        v16bf __attribute__((ext_vector_type(16)));
typedef float         v8f   __attribute__((ext_vector_type(8)));
typedef float         v4f   __attribute__((ext_vector_type(4)));
typedef unsigned int  v4u   __attribute__((ext_vector_type(4)));
typedef unsigned int  u32x4 __attribute__((ext_vector_type(4)));
typedef int           i32x8 __attribute__((ext_vector_type(8)));
typedef int           i32x4 __attribute__((ext_vector_type(4)));

#define NN   8192   // N nodes
#define CC   512    // IN_C == OUT_C
#define BM   128    // block M tile
#define BK   32     // K step (bf16 WMMA K)
#define LDA  40     // LDS row stride in bf16 elems (80B: b128-aligned, conflict-free)
#define LDB  40

#if __has_builtin(__builtin_amdgcn_tensor_load_to_lds)
#define HAVE_TDM 1
#else
#define HAVE_TDM 0
#endif

__device__ __forceinline__ unsigned short f2bf(float f) {
    union { float f; unsigned u; } c; c.f = f;
    unsigned r = c.u + 0x7FFFu + ((c.u >> 16) & 1u);   // round-to-nearest-even
    return (unsigned short)(r >> 16);
}
__device__ __forceinline__ unsigned pack2(float a, float b) {
    return (unsigned)f2bf(a) | ((unsigned)f2bf(b) << 16);
}

union FragAB { v4u q[2]; v16bf v; };   // 32B: one 16x32 (A) or 32x16 (B) bf16 fragment

// ---- Tensor Data Mover: 2D bf16 tile (tile_d0=32 contiguous elems = 64B rows),
// ---- LDS-padded to 80B rows (pad_interval=3 -> every 16 DW, pad_amount=3 -> 4 DW).
__device__ __forceinline__ void tdm_load_tile_bf16(const unsigned short* gptr,
                                                   unsigned lds_off_bytes,
                                                   unsigned tensor_d0, unsigned tensor_d1,
                                                   unsigned long long d0_stride_elems,
                                                   unsigned tile_d1) {
#if HAVE_TDM
    unsigned long long ga = (unsigned long long)(uintptr_t)gptr;
    u32x4 g0;
    g0.x = 0x1u;                                              // count=1 (valid user D#)
    g0.y = lds_off_bytes;                                     // lds_addr [63:32]
    g0.z = (unsigned)(ga & 0xFFFFFFFFu);                      // global_addr[31:0]
    g0.w = (unsigned)((ga >> 32) & 0x1FFFFFFu) | (2u << 30);  // addr[56:32] | type=2
    i32x8 g1;
    g1[0] = (int)((1u << 16)        // data_size = 1 -> 2 bytes
                | (1u << 20)        // pad_enable
                | (3u << 22)        // pad_interval: 16 DWORDs (64B) before padding
                | (3u << 25));      // pad_amount:   4 DWORDs (16B) of padding
    g1[1] = (int)((tensor_d0 & 0xFFFFu) << 16);                                   // [63:48] dim0 lo
    g1[2] = (int)(((tensor_d0 >> 16) & 0xFFFFu) | ((tensor_d1 & 0xFFFFu) << 16)); // dim0 hi | dim1 lo
    g1[3] = (int)(((tensor_d1 >> 16) & 0xFFFFu) | (32u << 16));                   // dim1 hi | tile_d0=32
    g1[4] = (int)(tile_d1 & 0xFFFFu);                                             // tile_d1 | tile_d2=0
    g1[5] = (int)(unsigned)(d0_stride_elems & 0xFFFFFFFFu);                       // dim0_stride lo32
    g1[6] = (int)(unsigned)((d0_stride_elems >> 32) & 0xFFFFu);                   // dim0_stride hi16
    g1[7] = 0;
    i32x4 z4 = {0, 0, 0, 0};
#if __clang_major__ >= 23
    i32x8 z8 = {0, 0, 0, 0, 0, 0, 0, 0};
    __builtin_amdgcn_tensor_load_to_lds(g0, g1, z4, z4, z8, 0);
#else
    __builtin_amdgcn_tensor_load_to_lds(g0, g1, z4, z4, 0);
#endif
#endif
}

// ---------------- kernel 1: dinv[i] = rsqrt(1 + sum_j adj[i][j]) ----------------
__global__ __launch_bounds__(256) void deg_kernel(const float* __restrict__ adj,
                                                  float* __restrict__ dinv) {
    const size_t row = blockIdx.x;
    const v4f* p = (const v4f*)(adj + row * (size_t)NN);
    float s = 0.f;
    for (int i = threadIdx.x; i < NN / 4; i += 256) {
        v4f v = p[i];
        s += (v.x + v.y) + (v.z + v.w);
    }
    #pragma unroll
    for (int off = 16; off; off >>= 1) s += __shfl_xor(s, off, 32);
    __shared__ float red[8];
    if ((threadIdx.x & 31) == 0) red[threadIdx.x >> 5] = s;
    __syncthreads();
    if (threadIdx.x == 0) {
        float t = 1.0f;                    // + identity diagonal
        #pragma unroll
        for (int i = 0; i < 8; i++) t += red[i];
        dinv[row] = rsqrtf(t);
    }
}

// ------- kernel 2: xs_t[c][j] = bf16(dinv[j] * x[j][c])   (transposed, n-major) -------
__global__ __launch_bounds__(256) void xst_kernel(const float* __restrict__ x,
                                                  const float* __restrict__ dinv,
                                                  unsigned short* __restrict__ xs_t) {
    const int j0 = blockIdx.x * 32;
    __shared__ float dsh[32];
    if (threadIdx.x < 32) dsh[threadIdx.x] = dinv[j0 + threadIdx.x];
    __syncthreads();
    #pragma unroll
    for (int cc = 0; cc < 2; cc++) {
        const int c = threadIdx.x + cc * 256;
        float v[32];
        #pragma unroll
        for (int i = 0; i < 32; i++) v[i] = dsh[i] * x[(size_t)(j0 + i) * CC + c];
        #pragma unroll
        for (int q = 0; q < 4; q++) {
            v4u o;
            o.x = pack2(v[q*8+0], v[q*8+1]); o.y = pack2(v[q*8+2], v[q*8+3]);
            o.z = pack2(v[q*8+4], v[q*8+5]); o.w = pack2(v[q*8+6], v[q*8+7]);
            *(v4u*)&xs_t[(size_t)c * NN + j0 + q * 8] = o;
        }
    }
}

// ------- kernel 3: wbt[n][k] = bf16(w[k][n])   (transposed weight) -------
__global__ __launch_bounds__(256) void wbt_kernel(const float* __restrict__ w,
                                                  unsigned short* __restrict__ wbt) {
    const int k0 = blockIdx.x * 32;
    #pragma unroll
    for (int cc = 0; cc < 2; cc++) {
        const int n = threadIdx.x + cc * 256;
        float v[32];
        #pragma unroll
        for (int i = 0; i < 32; i++) v[i] = w[(size_t)(k0 + i) * CC + n];
        #pragma unroll
        for (int q = 0; q < 4; q++) {
            v4u o;
            o.x = pack2(v[q*8+0], v[q*8+1]); o.y = pack2(v[q*8+2], v[q*8+3]);
            o.z = pack2(v[q*8+4], v[q*8+5]); o.w = pack2(v[q*8+6], v[q*8+7]);
            *(v4u*)&wbt[(size_t)n * CC + k0 + q * 8] = o;
        }
    }
}

// ---------------- kernel 4: h = bf16( dinv[m]*(adj @ xs) + dinv[m]^2*x )  ----------------
// LDS (dynamic, base offset 0): As @0 (10240B), Bs0 @10240 (40960B), Bs1 @51200 (40960B)
#define G1_AS   0u
#define G1_BS0  10240u
#define G1_BS1  51200u
__global__ __launch_bounds__(256, 1) void gcn_agg_kernel(const float* __restrict__ adj,
                                                         const unsigned short* __restrict__ xs_t,
                                                         const float* __restrict__ dinv,
                                                         const float* __restrict__ x,
                                                         unsigned short* __restrict__ h) {
    extern __shared__ char smem[];
    unsigned short* As = (unsigned short*)(smem + G1_AS);

    const int tid  = threadIdx.x;
    const int lane = tid & 31;
    const int wave = tid >> 5;                 // 0..7
    const int wm   = (wave >> 2) * 64;
    const int wn   = (wave & 3) * 128;
    const int l15  = lane & 15;
    const int lh   = lane >> 4;
    const size_t m0 = (size_t)blockIdx.x * BM;

    v8f acc[4][8];
    #pragma unroll
    for (int i = 0; i < 4; i++)
        #pragma unroll
        for (int j = 0; j < 8; j++)
            #pragma unroll
            for (int e = 0; e < 8; e++) acc[i][j][e] = 0.0f;

    const int ar = tid >> 1, ah = tid & 1;     // A staging: row 0..127, 16-col half

    // ---- prologue: TDM B tile 0, A globals tile 0 ----
#if HAVE_TDM
    if (wave == 0) tdm_load_tile_bf16(xs_t + 0, G1_BS0, NN, CC, NN, CC);
#endif
    v4f a0, a1, a2, a3;
    {
        const float* ga = adj + (m0 + ar) * (size_t)NN + ah * 16;
        a0 = *(const v4f*)(ga + 0); a1 = *(const v4f*)(ga + 4);
        a2 = *(const v4f*)(ga + 8); a3 = *(const v4f*)(ga + 12);
    }

    const int T = NN / BK;
    for (int it = 0; it < T; ++it) {
        const int k0 = it * BK;
        const int kn = (it + 1 < T) ? k0 + BK : k0;
#if HAVE_TDM
        const unsigned cur_off = (it & 1) ? G1_BS1 : G1_BS0;
        const unsigned nxt_off = (it & 1) ? G1_BS0 : G1_BS1;
#else
        const unsigned cur_off = G1_BS0;
#endif
        __syncthreads();   // all waves done reading As & Bs[nxt] from previous iterations

        // ---- A: fp32 -> bf16, row-major into LDS ----
        {
            v4u p0, p1;
            p0.x = pack2(a0.x, a0.y); p0.y = pack2(a0.z, a0.w);
            p0.z = pack2(a1.x, a1.y); p0.w = pack2(a1.z, a1.w);
            p1.x = pack2(a2.x, a2.y); p1.y = pack2(a2.z, a2.w);
            p1.z = pack2(a3.x, a3.y); p1.w = pack2(a3.z, a3.w);
            *(v4u*)&As[ar * LDA + ah * 16]     = p0;
            *(v4u*)&As[ar * LDA + ah * 16 + 8] = p1;
        }
#if HAVE_TDM
        if (wave == 0) tdm_load_tile_bf16(xs_t + kn, nxt_off, NN, CC, NN, CC);
#endif
        // ---- pipeline: next A tile into registers; prefetch 2 tiles ahead ----
        {
            const float* ga = adj + (m0 + ar) * (size_t)NN + kn + ah * 16;
            __builtin_prefetch(ga + 2 * BK, 0, 1);
            a0 = *(const v4f*)(ga + 0); a1 = *(const v4f*)(ga + 4);
            a2 = *(const v4f*)(ga + 8); a3 = *(const v4f*)(ga + 12);
        }
#if HAVE_TDM
        if (wave == 0) __builtin_amdgcn_s_wait_tensorcnt(1);   // current B tile complete
#else
        {   // fallback: cooperative copy of B tile k0 into Bs0
            unsigned short* Bs = (unsigned short*)(smem + G1_BS0);
            const int rn = tid * 2;
            #pragma unroll
            for (int rr = 0; rr < 2; rr++) {
                const unsigned short* src = xs_t + (size_t)(rn + rr) * NN + k0;
                unsigned short* dst = Bs + (rn + rr) * LDB;
                *(v4u*)(dst)      = *(const v4u*)(src);
                *(v4u*)(dst + 8)  = *(const v4u*)(src + 8);
                *(v4u*)(dst + 16) = *(const v4u*)(src + 16);
                *(v4u*)(dst + 24) = *(const v4u*)(src + 24);
            }
        }
#endif
        __syncthreads();

        // ---- fragments + WMMA ----
        const unsigned short* Bs = (const unsigned short*)(smem + cur_off);
        FragAB af[4];
        #pragma unroll
        for (int mt = 0; mt < 4; mt++) {
            const unsigned short* ap = &As[(wm + mt * 16 + l15) * LDA + lh * 8];
            af[mt].q[0] = *(const v4u*)(ap);         // K = lh*8 .. lh*8+7
            af[mt].q[1] = *(const v4u*)(ap + 16);    // K = lh*8+16 .. lh*8+23
        }
        #pragma unroll
        for (int nt = 0; nt < 8; nt++) {
            FragAB bf_;
            const unsigned short* bp = &Bs[(wn + nt * 16 + l15) * LDB + lh * 8];
            bf_.q[0] = *(const v4u*)(bp);
            bf_.q[1] = *(const v4u*)(bp + 16);
            #pragma unroll
            for (int mt = 0; mt < 4; mt++) {
                acc[mt][nt] = __builtin_amdgcn_wmma_f32_16x16x32_bf16(
                    false, af[mt].v, false, bf_.v, (short)0, acc[mt][nt], false, false);
            }
        }
    }

    // ---- epilogue: h = bf16( dinv[m]*acc + dinv[m]^2 * x[m][n] ) ----
    #pragma unroll
    for (int mt = 0; mt < 4; mt++) {
        #pragma unroll
        for (int e = 0; e < 8; e++) {
            const int m = (int)m0 + wm + mt * 16 + lh * 8 + e;
            const float dm = dinv[m];
            const size_t rb = (size_t)m * CC;
            #pragma unroll
            for (int nt = 0; nt < 8; nt++) {
                const int n = wn + nt * 16 + l15;
                float v = dm * (acc[mt][nt][e] + dm * x[rb + n]);
                h[rb + n] = f2bf(v);
            }
        }
    }
}

// ---------------- kernel 5: out = elu(h @ w)  [M=8192,K=512,N=512] ----------------
// LDS (dynamic): As0 @0, As1 @10240, Bs0 @20480, Bs1 @61440  (total 102400B)
#define G2_AS0  0u
#define G2_AS1  10240u
#define G2_BS0  20480u
#define G2_BS1  61440u
__global__ __launch_bounds__(256, 1) void gcn_lin_kernel(const unsigned short* __restrict__ hbuf,
                                                         const unsigned short* __restrict__ wbt,
                                                         float* __restrict__ out) {
    extern __shared__ char smem[];

    const int tid  = threadIdx.x;
    const int lane = tid & 31;
    const int wave = tid >> 5;
    const int wm   = (wave >> 2) * 64;
    const int wn   = (wave & 3) * 128;
    const int l15  = lane & 15;
    const int lh   = lane >> 4;
    const size_t m0 = (size_t)blockIdx.x * BM;

    v8f acc[4][8];
    #pragma unroll
    for (int i = 0; i < 4; i++)
        #pragma unroll
        for (int j = 0; j < 8; j++)
            #pragma unroll
            for (int e = 0; e < 8; e++) acc[i][j][e] = 0.0f;

#if HAVE_TDM
    if (wave == 0) {
        tdm_load_tile_bf16(hbuf + m0 * CC, G2_AS0, CC, NN, CC, BM);  // A tile 0: 128x32
        tdm_load_tile_bf16(wbt + 0,        G2_BS0, CC, CC, CC, CC);  // B tile 0: 512x32
    }
#endif

    const int T = CC / BK;
    for (int it = 0; it < T; ++it) {
        const int k0 = it * BK;
        const int kn = (it + 1 < T) ? k0 + BK : k0;
#if HAVE_TDM
        const unsigned a_cur = (it & 1) ? G2_AS1 : G2_AS0;
        const unsigned a_nxt = (it & 1) ? G2_AS0 : G2_AS1;
        const unsigned b_cur = (it & 1) ? G2_BS1 : G2_BS0;
        const unsigned b_nxt = (it & 1) ? G2_BS0 : G2_BS1;
#else
        const unsigned a_cur = G2_AS0, b_cur = G2_BS0;
#endif
        __syncthreads();
#if HAVE_TDM
        if (wave == 0) {
            tdm_load_tile_bf16(hbuf + m0 * CC + kn, a_nxt, CC, NN, CC, BM);
            tdm_load_tile_bf16(wbt + kn,            b_nxt, CC, CC, CC, CC);
            __builtin_amdgcn_s_wait_tensorcnt(2);   // current A+B tiles complete
        }
#else
        {   // fallback cooperative copies of tile k0
            unsigned short* Asw = (unsigned short*)(smem + G2_AS0);
            const int r = tid >> 1, hh = tid & 1;
            const unsigned short* srcA = hbuf + (m0 + r) * (size_t)CC + k0 + hh * 16;
            *(v4u*)&Asw[r * LDA + hh * 16]     = *(const v4u*)(srcA);
            *(v4u*)&Asw[r * LDA + hh * 16 + 8] = *(const v4u*)(srcA + 8);
            unsigned short* Bsw = (unsigned short*)(smem + G2_BS0);
            const int rn = tid * 2;
            #pragma unroll
            for (int rr = 0; rr < 2; rr++) {
                const unsigned short* src = wbt + (size_t)(rn + rr) * CC + k0;
                unsigned short* dst = Bsw + (rn + rr) * LDB;
                *(v4u*)(dst)      = *(const v4u*)(src);
                *(v4u*)(dst + 8)  = *(const v4u*)(src + 8);
                *(v4u*)(dst + 16) = *(const v4u*)(src + 16);
                *(v4u*)(dst + 24) = *(const v4u*)(src + 24);
            }
        }
#endif
        __syncthreads();

        const unsigned short* As = (const unsigned short*)(smem + a_cur);
        const unsigned short* Bs = (const unsigned short*)(smem + b_cur);
        FragAB af[4];
        #pragma unroll
        for (int mt = 0; mt < 4; mt++) {
            const unsigned short* ap = &As[(wm + mt * 16 + l15) * LDA + lh * 8];
            af[mt].q[0] = *(const v4u*)(ap);
            af[mt].q[1] = *(const v4u*)(ap + 16);
        }
        #pragma unroll
        for (int nt = 0; nt < 8; nt++) {
            FragAB bf_;
            const unsigned short* bp = &Bs[(wn + nt * 16 + l15) * LDB + lh * 8];
            bf_.q[0] = *(const v4u*)(bp);
            bf_.q[1] = *(const v4u*)(bp + 16);
            #pragma unroll
            for (int mt = 0; mt < 4; mt++) {
                acc[mt][nt] = __builtin_amdgcn_wmma_f32_16x16x32_bf16(
                    false, af[mt].v, false, bf_.v, (short)0, acc[mt][nt], false, false);
            }
        }
    }

    // ---- epilogue: ELU, fp32 out ----
    #pragma unroll
    for (int mt = 0; mt < 4; mt++) {
        #pragma unroll
        for (int e = 0; e < 8; e++) {
            const int m = (int)m0 + wm + mt * 16 + lh * 8 + e;
            const size_t rb = (size_t)m * CC;
            #pragma unroll
            for (int nt = 0; nt < 8; nt++) {
                const int n = wn + nt * 16 + l15;
                float v = acc[mt][nt][e];
                out[rb + n] = (v > 0.0f) ? v : (__expf(v) - 1.0f);
            }
        }
    }
}

extern "C" void kernel_launch(void* const* d_in, const int* in_sizes, int n_in,
                              void* d_out, int out_size, void* d_ws, size_t ws_size,
                              hipStream_t stream) {
    const float* x   = (const float*)d_in[0];   // [8192, 512]
    const float* adj = (const float*)d_in[1];   // [8192, 8192]
    const float* w   = (const float*)d_in[2];   // [512, 512]
    float* out = (float*)d_out;                 // [8192, 512] fp32

    char* ws = (char*)d_ws;
    float*          dinv = (float*)ws;                                          // 32 KB
    unsigned short* xst  = (unsigned short*)(ws + 32768);                       // 8 MB bf16 [512][8192]
    unsigned short* hb   = (unsigned short*)(ws + 32768 + 8u * 1024 * 1024);    // 8 MB bf16 [8192][512]
    unsigned short* wbt  = (unsigned short*)(ws + 32768 + 16u * 1024 * 1024);   // 512 KB bf16 [512][512]

    deg_kernel<<<NN, 256, 0, stream>>>(adj, dinv);
    xst_kernel<<<NN / 32, 256, 0, stream>>>(x, dinv, xst);
    wbt_kernel<<<CC / 32, 256, 0, stream>>>(w, wbt);
    gcn_agg_kernel<<<NN / BM, 256, 92160, stream>>>(adj, xst, dinv, x, hb);
    gcn_lin_kernel<<<NN / BM, 256, 102400, stream>>>(hb, wbt, out);
}
